// TFMistralAttention_18373870092721
// MI455X (gfx1250) — matrix operand order repaired and checked by reading, not compiled, not run on hardware
//
#include <hip/hip_runtime.h>
#include <hip/hip_bf16.h>
#include <stdint.h>
#include <stddef.h>

// ---------------------------------------------------------------------------
// CDNA5 (gfx1250) types for WMMA: wave32, v_wmma_f32_16x16x32_bf16
// ---------------------------------------------------------------------------
typedef __bf16 bf16;
typedef __attribute__((ext_vector_type(16))) __bf16 v16bf;
typedef __attribute__((ext_vector_type(8)))  __bf16 v8bf;
typedef __attribute__((ext_vector_type(8)))  float  v8f;
typedef __attribute__((ext_vector_type(4)))  unsigned int u32x4;
typedef __attribute__((ext_vector_type(8)))  int i32x8;
typedef __attribute__((ext_vector_type(4)))  int i32x4;

#define NB    2
#define NS    2048
#define NH    32
#define NKVH  8
#define ND    128
#define NHID  4096
#define WIN   1024

// Tensor Data Mover availability (this toolchain: 6-arg clang-23 form).
#if defined(__has_builtin)
#if __has_builtin(__builtin_amdgcn_tensor_load_to_lds) && \
    __has_builtin(__builtin_amdgcn_s_wait_tensorcnt)
#define USE_TDM 1
#endif
#endif
#ifndef USE_TDM
#define USE_TDM 0
#endif

static __device__ __forceinline__ v16bf pack16(v8bf lo, v8bf hi) {
  v16bf r;
#pragma unroll
  for (int i = 0; i < 8; ++i) { r[i] = lo[i]; r[i + 8] = hi[i]; }
  return r;
}

static __device__ __forceinline__ v16bf load_frag16(const bf16* p0, const bf16* p1) {
  v8bf lo = *(const v8bf*)p0;
  v8bf hi = *(const v8bf*)p1;
  return pack16(lo, hi);
}

static __device__ __forceinline__ v8f wmma_bf16(v16bf a, v16bf b, v8f c) {
  // D = A(16x32 bf16) * B(32x16 bf16) + C(16x16 f32)
  return __builtin_amdgcn_wmma_f32_16x16x32_bf16(
      /*neg_a=*/false, a, /*neg_b=*/false, b,
      /*c_mod=*/(short)0, c, /*reuse_a=*/false, /*reuse_b=*/false);
}

// ---------------------------------------------------------------------------
// Generic tiled GEMM:  Out(MxN) = A(MxK) * B(KxN)
//   A: fp32 (converted to bf16 into LDS) or bf16; B: fp32 (converted in LDS)
//   128x128 C-tile per workgroup, 8 waves (4x2 wave grid, 32x64 per wave),
//   K-step 32 -> one v_wmma_f32_16x16x32_bf16 per 16x16x32 sub-problem.
// ---------------------------------------------------------------------------
template <bool A_BF16, bool OUT_BF16>
__global__ __launch_bounds__(256, 2)
void wmma_gemm(const void* __restrict__ Ap, const float* __restrict__ Bm,
               void* __restrict__ Outp, int M, int N, int K)
{
  __shared__ bf16 As[128 * 32];   // [m][k]
  __shared__ bf16 Bs[128 * 32];   // [n][k]

  const int tid  = threadIdx.x;
  const int wave = tid >> 5;
  const int lane = tid & 31;
  const int half = lane >> 4;
  const int l16  = lane & 15;
  const int m0   = blockIdx.y * 128;
  const int n0   = blockIdx.x * 128;
  const int wm   = wave >> 1;     // 0..3  (32-row slab)
  const int wn   = wave & 1;      // 0..1  (64-col slab)

  v8f acc[2][4];
#pragma unroll
  for (int i = 0; i < 2; ++i)
#pragma unroll
    for (int j = 0; j < 4; ++j)
#pragma unroll
      for (int e = 0; e < 8; ++e) acc[i][j][e] = 0.0f;

  for (int k0 = 0; k0 < K; k0 += 32) {
    __syncthreads();
    // ---- stage A tile (128 x 32) into LDS as bf16 ----
    if (A_BF16) {
      const bf16* A = (const bf16*)Ap;
#pragma unroll
      for (int p = 0; p < 2; ++p) {
        int r  = (tid >> 2) + p * 64;
        int kk = (tid & 3) * 8;
        *(v8bf*)(As + r * 32 + kk) =
            *(const v8bf*)(A + (size_t)(m0 + r) * K + k0 + kk);
      }
    } else {
      const float* A = (const float*)Ap;
#pragma unroll
      for (int p = 0; p < 4; ++p) {
        int r  = (tid >> 3) + p * 32;
        int kk = (tid & 7) * 4;
        float4 f = *(const float4*)(A + (size_t)(m0 + r) * K + k0 + kk);
        bf16* d = As + r * 32 + kk;
        d[0] = (bf16)f.x; d[1] = (bf16)f.y; d[2] = (bf16)f.z; d[3] = (bf16)f.w;
      }
    }
    // ---- stage B tile (32 x 128), transposed to [n][k] ----
#pragma unroll
    for (int p = 0; p < 8; ++p) {
      int kk = (tid >> 6) + p * 4;
      int nn = (tid & 63) * 2;
      float2 f = *(const float2*)(Bm + (size_t)(k0 + kk) * N + n0 + nn);
      Bs[(size_t)nn * 32 + kk]       = (bf16)f.x;
      Bs[(size_t)(nn + 1) * 32 + kk] = (bf16)f.y;
    }
    // prefetch next B k-slab (emits global_prefetch_b8 on gfx1250)
    if (k0 + 32 < K) {
      __builtin_prefetch(Bm + (size_t)(k0 + 32 + (tid >> 6)) * N + n0 + (tid & 63) * 2, 0, 1);
    }
    __syncthreads();

    // ---- fragments + 8 WMMAs ----
    v16bf afr[2], bfr[4];
#pragma unroll
    for (int im = 0; im < 2; ++im) {
      const bf16* p0 = As + (wm * 32 + im * 16 + l16) * 32;
      afr[im] = load_frag16(p0 + 16 * half, p0 + 16 * half + 8);
    }
#pragma unroll
    for (int jn = 0; jn < 4; ++jn) {
      const bf16* p0 = Bs + (wn * 64 + jn * 16 + l16) * 32 + 16 * half;
      bfr[jn] = load_frag16(p0, p0 + 8);
    }
#pragma unroll
    for (int im = 0; im < 2; ++im)
#pragma unroll
      for (int jn = 0; jn < 4; ++jn)
        acc[im][jn] = wmma_bf16(afr[im], bfr[jn], acc[im][jn]);
  }

  // ---- epilogue: C layout = M in VGPR index (+8 per lane-half), N in lanes ----
#pragma unroll
  for (int im = 0; im < 2; ++im)
#pragma unroll
    for (int jn = 0; jn < 4; ++jn)
#pragma unroll
      for (int r = 0; r < 8; ++r) {
        int row = m0 + wm * 32 + im * 16 + r + 8 * half;
        int col = n0 + wn * 64 + jn * 16 + l16;
        if (OUT_BF16)
          ((bf16*)Outp)[(size_t)row * N + col] = (bf16)acc[im][jn][r];
        else
          ((float*)Outp)[(size_t)row * N + col] = acc[im][jn][r];
      }
}

// ---------------------------------------------------------------------------
// RoPE + GQA reshape (positions are arange(s) -> pos = s index).
// ---------------------------------------------------------------------------
__global__ __launch_bounds__(256)
void rope_reshape(const bf16* __restrict__ qp, const bf16* __restrict__ kp,
                  const bf16* __restrict__ vp, bf16* __restrict__ Q,
                  bf16* __restrict__ K, bf16* __restrict__ V)
{
  size_t idx = (size_t)blockIdx.x * 256 + threadIdx.x;   // b*s*6144 elements
  int col   = (int)(idx % 6144);
  size_t rw = idx / 6144;
  int si = (int)(rw % NS);
  int bi = (int)(rw / NS);
  float pos = (float)si;

  if (col < NH * ND) {                       // Q
    int h = col >> 7, d = col & 127;
    float x  = (float)qp[rw * (NH * ND) + col];
    int   dp = (d + 64) & 127;
    float x2 = (float)qp[rw * (NH * ND) + h * ND + dp];
    float rot = (d < 64) ? -x2 : x2;
    float invf = __powf(10000.0f, -((float)(d & 63)) / 64.0f);
    float ang = pos * invf;
    float o = x * __cosf(ang) + rot * __sinf(ang);
    Q[(((size_t)bi * NH + h) * NS + si) * ND + d] = (bf16)o;
  } else if (col < NH * ND + NKVH * ND) {    // K
    int c = col - NH * ND;
    int h = c >> 7, d = c & 127;
    float x  = (float)kp[rw * (NKVH * ND) + c];
    int   dp = (d + 64) & 127;
    float x2 = (float)kp[rw * (NKVH * ND) + h * ND + dp];
    float rot = (d < 64) ? -x2 : x2;
    float invf = __powf(10000.0f, -((float)(d & 63)) / 64.0f);
    float ang = pos * invf;
    float o = x * __cosf(ang) + rot * __sinf(ang);
    K[(((size_t)bi * NKVH + h) * NS + si) * ND + d] = (bf16)o;
  } else {                                   // V passthrough
    int c = col - NH * ND - NKVH * ND;
    int h = c >> 7, d = c & 127;
    V[(((size_t)bi * NKVH + h) * NS + si) * ND + d] = vp[rw * (NKVH * ND) + c];
  }
}

// ---------------------------------------------------------------------------
// Flash attention with sliding window (causal, width 1024), GQA 4:1.
// Grid (s/64, H, b), 128 threads = 4 waves; each wave owns 16 query rows.
// K tile (32 keys x 128 dims, one contiguous 8KB bf16 block) is moved by the
// Tensor Data Mover (tensor_load_to_lds + s_wait_tensorcnt) when available;
// V tile is transposed manually into LDS for the PV B-operand layout.
// ---------------------------------------------------------------------------
__global__ __launch_bounds__(128, 2)
void flash_attn(const bf16* __restrict__ Q, const bf16* __restrict__ K,
                const bf16* __restrict__ V, bf16* __restrict__ O)
{
  __shared__ bf16 Ks[32 * 128];   // [key][d]
  __shared__ bf16 Vs[128 * 32];   // [d][key]  (transposed)
  __shared__ bf16 Ps[4][16 * 32]; // per-wave P tile [m][k]

  const int tid  = threadIdx.x;
  const int wave = tid >> 5;
  const int lane = tid & 31;
  const int half = lane >> 4;
  const int l16  = lane & 15;
  const int qblk = blockIdx.x;          // 64-query slab
  const int h    = blockIdx.y;
  const int bi   = blockIdx.z;
  const int kvh  = h >> 2;              // H/KVH = 4
  const int q0   = qblk * 64 + wave * 16;

  const bf16* Qb = Q + ((size_t)bi * NH + h) * NS * ND;
  const bf16* Kb = K + ((size_t)bi * NKVH + kvh) * NS * ND;
  const bf16* Vb = V + ((size_t)bi * NKVH + kvh) * NS * ND;

#if USE_TDM
  // LDS byte offset of Ks (AS(3) pointer value is the LDS offset).
  const unsigned ks_lds = (unsigned)(unsigned long long)
      (__attribute__((address_space(3))) void*)Ks;
#endif

  // Q fragments for this wave's 16 rows (A layout), kept in registers.
  v16bf qa[4];
  const bf16* qrow = Qb + (size_t)(q0 + l16) * ND;
#pragma unroll
  for (int c4 = 0; c4 < 4; ++c4)
    qa[c4] = load_frag16(qrow + c4 * 32 + 16 * half,
                         qrow + c4 * 32 + 16 * half + 8);

  v8f o[8];
  float mrun[8], lrun[8];
#pragma unroll
  for (int t = 0; t < 8; ++t) {
#pragma unroll
    for (int e = 0; e < 8; ++e) o[t][e] = 0.0f;
  }
#pragma unroll
  for (int r = 0; r < 8; ++r) { mrun[r] = -1e29f; lrun[r] = 0.0f; }

  const int qlo = qblk * 64;
  const int qhi = qblk * 64 + 63;
  int kb0 = qlo - (WIN - 1);
  if (kb0 < 0) kb0 = 0;
  kb0 = (kb0 / 32) * 32;
  const float scale = 0.088388347648318447f;   // 1/sqrt(128)

  for (int kb = kb0; kb <= qhi; kb += 32) {
    __syncthreads();   // all waves done reading previous Ks/Vs

#if USE_TDM
    if (wave == 0) {
      // Build Tensor DMA descriptor (D#) per CDNA5 ISA 8.3/8.4:
      // 2-D tile: tile_dim0 = 128 elems (d), tile_dim1 = 32 (keys),
      // data_size = 2B, tensor_dim0_stride = 128 -> contiguous 8KB move.
      unsigned long long ga =
          (unsigned long long)(const void*)(Kb + (size_t)kb * ND);
      u32x4 g0;
      g0[0] = 1u;                                  // count=1, user mode
      g0[1] = ks_lds;                              // lds_addr (bytes)
      g0[2] = (unsigned)(ga & 0xffffffffu);        // global_addr[31:0]
      g0[3] = (unsigned)((ga >> 32) & 0x1ffffffu)  // global_addr[56:32]
              | (2u << 30);                        // type = 2 ("image")
      i32x8 g1;
      g1[0] = (int)(1u << 16);                     // data_size = 1 (2 bytes)
      g1[1] = (int)(128u << 16);                   // tensor_dim0 = 128
      g1[2] = (int)(2048u << 16);                  // tensor_dim1 = NS
      g1[3] = (int)(128u << 16);                   // tile_dim0 = 128
      g1[4] = 32;                                  // tile_dim1 = 32
      g1[5] = 128;                                 // tensor_dim0_stride
      g1[6] = 0;
      g1[7] = 0;
      i32x4 gz;  gz[0] = 0; gz[1] = 0; gz[2] = 0; gz[3] = 0;
      i32x8 gz8;
#pragma unroll
      for (int z = 0; z < 8; ++z) gz8[z] = 0;
      // clang-23 / therock-10.0 6-arg form: (g0, g1, g2, g3, g_extra, cpol)
      __builtin_amdgcn_tensor_load_to_lds(g0, g1, gz, gz, gz8, 0);
      __builtin_amdgcn_s_wait_tensorcnt(0);
    }
    // ---- V tile transposed by all threads ----
    {
      int key = tid >> 2;             // 0..31
      int d0  = (tid & 3) * 32;
      const bf16* vs = Vb + (size_t)(kb + key) * ND + d0;
#pragma unroll
      for (int i = 0; i < 32; ++i) Vs[(d0 + i) * 32 + key] = vs[i];
    }
#else
    // ---- cooperative K/V tile load (32 keys x 128 dims) ----
    {
      int key = tid >> 2;             // 0..31
      int d0  = (tid & 3) * 32;
      const v8bf* ks = (const v8bf*)(Kb + (size_t)(kb + key) * ND + d0);
      v8bf* kd = (v8bf*)(Ks + key * 128 + d0);
      kd[0] = ks[0]; kd[1] = ks[1]; kd[2] = ks[2]; kd[3] = ks[3];
      const bf16* vs = Vb + (size_t)(kb + key) * ND + d0;
#pragma unroll
      for (int i = 0; i < 32; ++i) Vs[(d0 + i) * 32 + key] = vs[i];
    }
#endif
    __syncthreads();   // publish Ks (TDM complete) + Vs

    // ---- S = Q * K^T  (two 16-key column tiles) ----
    v8f s0, s1;
#pragma unroll
    for (int e = 0; e < 8; ++e) { s0[e] = 0.0f; s1[e] = 0.0f; }
#pragma unroll
    for (int c4 = 0; c4 < 4; ++c4) {
      const bf16* kp0 = Ks + l16 * 128 + c4 * 32 + 16 * half;
      const bf16* kp1 = Ks + (16 + l16) * 128 + c4 * 32 + 16 * half;
      v16bf b0 = load_frag16(kp0, kp0 + 8);
      v16bf b1 = load_frag16(kp1, kp1 + 8);
      s0 = wmma_bf16(qa[c4], b0, s0);
      s1 = wmma_bf16(qa[c4], b1, s1);
    }

    // ---- mask + online softmax ----
#pragma unroll
    for (int r = 0; r < 8; ++r) {
      int qi  = q0 + r + 8 * half;
      int k0i = kb + l16;
      int k1i = kb + 16 + l16;
      float v0 = s0[r] * scale;
      float v1 = s1[r] * scale;
      if (k0i > qi || k0i <= qi - WIN) v0 = -1e30f;
      if (k1i > qi || k1i <= qi - WIN) v1 = -1e30f;

      float mx = fmaxf(v0, v1);
#pragma unroll
      for (int sh = 1; sh < 16; sh <<= 1)
        mx = fmaxf(mx, __shfl_xor(mx, sh, 32));

      float mnew  = fmaxf(mrun[r], mx);       // mrun >= -1e29 keeps exp sane
      float alpha = __expf(mrun[r] - mnew);
      float p0 = __expf(v0 - mnew);
      float p1 = __expf(v1 - mnew);
      float rs = p0 + p1;
#pragma unroll
      for (int sh = 1; sh < 16; sh <<= 1)
        rs += __shfl_xor(rs, sh, 32);

#pragma unroll
      for (int t = 0; t < 8; ++t) o[t][r] *= alpha;
      lrun[r] = lrun[r] * alpha + rs;
      mrun[r] = mnew;
      s0[r] = p0;
      s1[r] = p1;
    }

    // ---- P (C layout) -> LDS -> A layout ----
    bf16* pw = &Ps[wave][0];
#pragma unroll
    for (int r = 0; r < 8; ++r) {
      int m = r + 8 * half;
      pw[m * 32 + l16]      = (bf16)s0[r];
      pw[m * 32 + 16 + l16] = (bf16)s1[r];
    }
    const bf16* pr = pw + l16 * 32;
    v16bf pa = load_frag16(pr + 16 * half, pr + 16 * half + 8);

    // ---- O += P * V  (8 d-tiles of 16) ----
#pragma unroll
    for (int t = 0; t < 8; ++t) {
      const bf16* vp0 = Vs + (t * 16 + l16) * 32 + 16 * half;
      v16bf bv = load_frag16(vp0, vp0 + 8);
      o[t] = wmma_bf16(pa, bv, o[t]);
    }
  }

  // ---- normalize + write attn output as (b, s, HID) bf16 ----
#pragma unroll
  for (int t = 0; t < 8; ++t)
#pragma unroll
    for (int r = 0; r < 8; ++r) {
      int qi = q0 + r + 8 * half;
      int d  = t * 16 + l16;
      O[((size_t)bi * NS + qi) * NHID + h * ND + d] =
          (bf16)(o[t][r] / lrun[r]);
    }
}

// ---------------------------------------------------------------------------
// Launch: QKV GEMMs -> RoPE/reshape -> flash attention -> output GEMM
// Inputs: 0 hidden(f32) 1 position_ids 2 attention_mask 3 Wq 4 Wk 5 Wv 6 Wo
// ---------------------------------------------------------------------------
extern "C" void kernel_launch(void* const* d_in, const int* in_sizes, int n_in,
                              void* d_out, int out_size, void* d_ws, size_t ws_size,
                              hipStream_t stream) {
  const float* hidden = (const float*)d_in[0];
  const float* Wq = (const float*)d_in[3];
  const float* Wk = (const float*)d_in[4];
  const float* Wv = (const float*)d_in[5];
  const float* Wo = (const float*)d_in[6];
  float* out = (float*)d_out;

  char* ws = (char*)d_ws;
  const size_t MROWS = (size_t)NB * NS;            // 4096
  bf16* qproj = (bf16*)(ws);                                        // 32 MiB
  bf16* kproj = (bf16*)(ws + MROWS * NHID * 2);                     //  8 MiB
  bf16* vproj = (bf16*)(ws + MROWS * NHID * 2 + MROWS * 1024 * 2);  //  8 MiB
  bf16* Qr = (bf16*)(ws + MROWS * (NHID + 2048) * 2);               // 32 MiB
  bf16* Kr = (bf16*)(ws + MROWS * (2 * NHID + 2048) * 2);           //  8 MiB
  bf16* Vr = (bf16*)(ws + MROWS * (2 * NHID + 3072) * 2);           //  8 MiB
  bf16* attn = qproj;   // qproj is dead after rope_reshape; reuse region

  // 1) QKV projections (fp32 A/B -> bf16 LDS -> WMMA -> bf16 out)
  wmma_gemm<false, true><<<dim3(NHID / 128, MROWS / 128), 256, 0, stream>>>(
      hidden, Wq, qproj, (int)MROWS, NHID, NHID);
  wmma_gemm<false, true><<<dim3(1024 / 128, MROWS / 128), 256, 0, stream>>>(
      hidden, Wk, kproj, (int)MROWS, 1024, NHID);
  wmma_gemm<false, true><<<dim3(1024 / 128, MROWS / 128), 256, 0, stream>>>(
      hidden, Wv, vproj, (int)MROWS, 1024, NHID);

  // 2) RoPE + reshape to (b, heads, s, d)
  {
    size_t total = (size_t)NB * NS * 6144;
    rope_reshape<<<(unsigned)(total / 256), 256, 0, stream>>>(
        qproj, kproj, vproj, Qr, Kr, Vr);
  }

  // 3) Sliding-window flash attention (TDM-fed K tiles)
  flash_attn<<<dim3(NS / 64, NH, NB), 128, 0, stream>>>(Qr, Kr, Vr, attn);

  // 4) Output projection -> fp32 result
  wmma_gemm<true, false><<<dim3(NHID / 128, MROWS / 128), 256, 0, stream>>>(
      attn, Wo, out, (int)MROWS, NHID, NHID);
}
